// Model_59407987638829
// MI455X (gfx1250) — matrix operand order repaired and checked by reading, not compiled, not run on hardware
//
#include <hip/hip_runtime.h>
#include <math.h>

// ---------------------------------------------------------------------------
// Liquid ODE-GRU scan for MI455X (gfx1250, wave32, WMMA bf16 16x16x32).
// B=256 rows split into 16 independent persistent workgroups of 16 rows each
// (the time recurrence never mixes batch rows -> no grid-wide sync needed).
// ---------------------------------------------------------------------------

#define HID     256
#define TSTEPS  64
#define BTOT    256
#define BTILE   16
#define THREADS 256   // 8 waves of 32

typedef __attribute__((ext_vector_type(16))) __bf16 v16bf;
typedef __attribute__((ext_vector_type(8)))  __bf16 v8bf;
typedef __attribute__((ext_vector_type(8)))  float  v8f;

// bf16 weight workspace layout (element offsets, 2B each)
static constexpr size_t OFF_ODE1 = 0;
static constexpr size_t OFF_ODE2 = OFF_ODE1 + 256 * 256;
static constexpr size_t OFF_ODE3 = OFF_ODE2 + 256 * 256;
static constexpr size_t OFF_GW1  = OFF_ODE3 + 256 * 256;
static constexpr size_t OFF_GW2  = OFF_GW1  + 256 * 256;
static constexpr size_t OFF_SW1  = OFF_GW2  + 256 * 256;
static constexpr size_t OFF_SW2  = OFF_SW1  + 256 * 256;
static constexpr size_t OFF_SW3  = OFF_SW2  + 256 * 256;   // 64 x 256
static constexpr size_t OFF_WIH  = OFF_SW3  + 64  * 256;   // 768 x 64 (K padded 63->64)
static constexpr size_t OFF_WHH  = OFF_WIH  + 768 * 64;    // 768 x 256

__device__ __forceinline__ unsigned short f2bf(float f) {
  unsigned u = __float_as_uint(f);
  unsigned r = (u + 0x7FFFu + ((u >> 16) & 1u)) >> 16;   // round-to-nearest-even
  return (unsigned short)r;
}
__device__ __forceinline__ float bf2f(unsigned short h) {
  return __uint_as_float(((unsigned)h) << 16);
}
__device__ __forceinline__ float gelu_f(float x) {
  return 0.5f * x * (1.0f + erff(x * 0.7071067811865476f));
}
__device__ __forceinline__ float sigm_f(float x) {
  return 1.0f / (1.0f + __expf(-x));
}

// A fragment (16xK activations, bf16, LDS). ISA 16-bit A layout:
// lane half 0 -> K {kb..kb+7, kb+16..kb+23}; half 1 -> {kb+8..15, kb+24..31}.
__device__ __forceinline__ v16bf fragA(const unsigned short* buf, int stride, int kb, int lane) {
  int m = lane & 15, half = lane >> 4;
  const unsigned short* p = buf + m * stride + kb + half * 8;
  v8bf lo = *(const v8bf*)(p);        // 16B -> ds_read_b128 pattern
  v8bf hi = *(const v8bf*)(p + 16);
  return __builtin_shufflevector(lo, hi, 0,1,2,3,4,5,6,7,8,9,10,11,12,13,14,15);
}

// B fragment (K x 16 slice of W^T): lane = N = row of W, same K order as A:
// half 0 -> K {kb..kb+7, kb+16..kb+23}; half 1 -> {kb+8..15, kb+24..31}.
__device__ __forceinline__ v16bf fragB(const unsigned short* W, int Kpad, int n0, int kb, int lane) {
  int n = n0 + (lane & 15), half = lane >> 4;
  const unsigned short* p = W + (size_t)n * Kpad + kb + half * 8;
  v8bf lo = *(const v8bf*)(p);
  v8bf hi = *(const v8bf*)(p + 16);
  return __builtin_shufflevector(lo, hi, 0,1,2,3,4,5,6,7,8,9,10,11,12,13,14,15);
}

__device__ __forceinline__ v8f wmma_bf(v16bf a, v16bf b, v8f c) {
  return __builtin_amdgcn_wmma_f32_16x16x32_bf16(false, a, false, b, (short)0, c, false, false);
}

// Accumulate a 16x32 output tile (2 WMMA tiles) over K.
__device__ __forceinline__ void mm_tile(const unsigned short* inB, int aStride, int K,
                                        const unsigned short* W, int Kpad, int n0,
                                        v8f& acc0, v8f& acc1, int lane) {
  __builtin_prefetch(W + (size_t)n0 * Kpad, 0, 1);
  for (int kb = 0; kb < K; kb += 32) {
    v16bf a  = fragA(inB, aStride, kb, lane);
    v16bf b0 = fragB(W, Kpad, n0,      kb, lane);
    v16bf b1 = fragB(W, Kpad, n0 + 16, kb, lane);
    acc0 = wmma_bf(a, b0, acc0);
    acc1 = wmma_bf(a, b1, acc1);
  }
}

// Full 256->256 layer: out = act(in @ W^T + bias), bf16 LDS staging.
__device__ __forceinline__ void layer256(const unsigned short* inB, const unsigned short* W,
                                         const float* bias, unsigned short* outB,
                                         int lane, int wave, bool act) {
  const int n0 = wave * 32;
  v8f acc0 = {}, acc1 = {};
  mm_tile(inB, HID, HID, W, HID, n0, acc0, acc1, lane);
  int n = lane & 15, rb = (lane >> 4) * 8;
  float bs0 = bias[n0 + n], bs1 = bias[n0 + 16 + n];
#pragma unroll
  for (int v = 0; v < 8; ++v) {
    int m = rb + v;
    float x0 = acc0[v] + bs0, x1 = acc1[v] + bs1;
    if (act) { x0 = gelu_f(x0); x1 = gelu_f(x1); }
    outB[m * HID + n0 + n]      = f2bf(x0);
    outB[m * HID + n0 + 16 + n] = f2bf(x1);
  }
}

struct Params {
  const float *inputs, *mask, *b_ih, *b_hh;
  const float *ode_b1, *ode_b2, *ode_b3;
  const float *g_b1, *g_b2, *g_w3, *g_b3;
  const float *s_b1, *s_b2, *s_b3;
  const unsigned short* wts;
  float *out_pred, *out_state;
};

__global__ void __launch_bounds__(THREADS, 1) convert_bf16_pad(const float* src, unsigned short* dst,
                                                               int rows, int K, int Kpad) {
  int i = blockIdx.x * THREADS + threadIdx.x;
  if (i >= rows * Kpad) return;
  int r = i / Kpad, c = i - r * Kpad;
  float v = (c < K) ? src[(size_t)r * K + c] : 0.0f;
  dst[i] = f2bf(v);
}

__global__ void __launch_bounds__(THREADS, 1) liquid_ode_gru(Params p) {
  __shared__ __align__(16) unsigned short aB[BTILE * HID];   // A-operand staging (bf16 of h / htmp)
  __shared__ __align__(16) unsigned short tB[BTILE * HID];   // layer staging
  __shared__ __align__(16) unsigned short cB[BTILE * HID];   // layer staging
  __shared__ __align__(16) unsigned short xB[BTILE * 64];    // GRU input (63 feats + 1 zero pad)
  __shared__ __align__(16) float hF[BTILE * HID];            // fp32 master hidden state
  __shared__ float predL[BTILE];
  __shared__ float dtS;

  const int tid  = threadIdx.x;
  const int lane = tid & 31;
  const int wave = tid >> 5;
  const int b0   = blockIdx.x * BTILE;

  const unsigned short* ODE1 = p.wts + OFF_ODE1;
  const unsigned short* ODE2 = p.wts + OFF_ODE2;
  const unsigned short* ODE3 = p.wts + OFF_ODE3;
  const unsigned short* GW1  = p.wts + OFF_GW1;
  const unsigned short* GW2  = p.wts + OFF_GW2;
  const unsigned short* SW1  = p.wts + OFF_SW1;
  const unsigned short* SW2  = p.wts + OFF_SW2;
  const unsigned short* SW3  = p.wts + OFF_SW3;
  const unsigned short* WIH  = p.wts + OFF_WIH;
  const unsigned short* WHH  = p.wts + OFF_WHH;

  for (int i = tid; i < BTILE * HID; i += THREADS) { hF[i] = 0.0f; aB[i] = 0; }
  __syncthreads();

#pragma unroll 1
  for (int t = 0; t < TSTEPS; ++t) {
    if (tid == 0) {
      float dt = 0.0f;
      if (t < TSTEPS - 1) {
        float t0 = p.inputs[(size_t)t * 64];
        float t1 = p.inputs[(size_t)(t + 1) * 64];
        if (isnan(t0)) t0 = 0.0f;
        if (isnan(t1)) t1 = 0.0f;
        dt = (t1 - t0) * 0.001f;
      }
      dtS = dt;
    }

    // ---- prediction head: pred = MLP3_g(h) ----
    layer256(aB, GW1, p.g_b1, tB, lane, wave, true); __syncthreads();
    layer256(tB, GW2, p.g_b2, cB, lane, wave, true); __syncthreads();
    if (tid < BTILE) {
      float s = p.g_b3[0];
      for (int k = 0; k < HID; ++k) s += bf2f(cB[tid * HID + k]) * p.g_w3[k];
      predL[tid] = s;
      if (t >= 1) p.out_pred[(size_t)(t - 1) * BTOT + b0 + tid] = s;
    }

    // ---- state head: state = MLP3_s(h), N=64 final layer ----
    layer256(aB, SW1, p.s_b1, tB, lane, wave, true); __syncthreads();
    layer256(tB, SW2, p.s_b2, cB, lane, wave, true); __syncthreads();
    if (wave < 4) {                       // wave-uniform branch: EXEC all-ones for WMMA
      const int n0 = wave * 16;
      v8f acc = {};
      for (int kb = 0; kb < HID; kb += 32)
        acc = wmma_bf(fragA(cB, HID, kb, lane), fragB(SW3, HID, n0, kb, lane), acc);
      if (t >= 1) {
        int n = lane & 15, rb = (lane >> 4) * 8;
        float b3 = p.s_b3[n0 + n];
#pragma unroll
        for (int v = 0; v < 8; ++v) {
          int m = rb + v;
          p.out_state[((size_t)(t - 1) * BTOT + b0 + m) * 64 + n0 + n] = acc[v] + b3;
        }
      }
    }
    __syncthreads();

    // ---- assemble GRU input: x = [blood, feats[:,1:]], zero-padded to 64 ----
    for (int i = tid; i < BTILE * 64; i += THREADS) {
      int m = i >> 6, c = i & 63;
      float val = 0.0f;
      if (c < 63) {
        float f = p.inputs[((size_t)(b0 + m) * TSTEPS + t) * 64 + 1 + c];
        if (isnan(f)) f = 0.0f;
        if (c == 0) {
          float mk = p.mask[(size_t)(b0 + m) * TSTEPS + t];
          val = mk * f + (1.0f - mk) * predL[m];
        } else {
          val = f;
        }
      }
      xB[i] = f2bf(val);
    }
    __syncthreads();

    // ---- GRU: r/z fuse x-part + h-part into one accumulator chain ----
    {
      const int n0 = wave * 32;
      v8f ar0 = {}, ar1 = {}, az0 = {}, az1 = {}, in0 = {}, in1 = {}, hn0 = {}, hn1 = {};
#pragma unroll
      for (int kb = 0; kb < 64; kb += 32) {     // x-part, K padded 63->64
        v16bf a = fragA(xB, 64, kb, lane);
        ar0 = wmma_bf(a, fragB(WIH, 64,           n0,      kb, lane), ar0);
        ar1 = wmma_bf(a, fragB(WIH, 64,           n0 + 16, kb, lane), ar1);
        az0 = wmma_bf(a, fragB(WIH, 64,     HID + n0,      kb, lane), az0);
        az1 = wmma_bf(a, fragB(WIH, 64,     HID + n0 + 16, kb, lane), az1);
        in0 = wmma_bf(a, fragB(WIH, 64, 2 * HID + n0,      kb, lane), in0);
        in1 = wmma_bf(a, fragB(WIH, 64, 2 * HID + n0 + 16, kb, lane), in1);
      }
#pragma unroll 1
      for (int kb = 0; kb < HID; kb += 32) {    // h-part
        v16bf a = fragA(aB, HID, kb, lane);
        ar0 = wmma_bf(a, fragB(WHH, HID,           n0,      kb, lane), ar0);
        ar1 = wmma_bf(a, fragB(WHH, HID,           n0 + 16, kb, lane), ar1);
        az0 = wmma_bf(a, fragB(WHH, HID,     HID + n0,      kb, lane), az0);
        az1 = wmma_bf(a, fragB(WHH, HID,     HID + n0 + 16, kb, lane), az1);
        hn0 = wmma_bf(a, fragB(WHH, HID, 2 * HID + n0,      kb, lane), hn0);
        hn1 = wmma_bf(a, fragB(WHH, HID, 2 * HID + n0 + 16, kb, lane), hn1);
      }
      __syncthreads();   // all waves done reading aB before rewriting it
      int n = lane & 15, rb = (lane >> 4) * 8;
#pragma unroll
      for (int tt = 0; tt < 2; ++tt) {
        int col   = n0 + tt * 16 + n;
        float brr = p.b_ih[col] + p.b_hh[col];
        float bzz = p.b_ih[HID + col] + p.b_hh[HID + col];
        float bin = p.b_ih[2 * HID + col];
        float bhn = p.b_hh[2 * HID + col];
        v8f& vr = tt ? ar1 : ar0;
        v8f& vz = tt ? az1 : az0;
        v8f& vi = tt ? in1 : in0;
        v8f& vh = tt ? hn1 : hn0;
#pragma unroll
        for (int v = 0; v < 8; ++v) {
          int m = rb + v;
          float r  = sigm_f(vr[v] + brr);
          float z  = sigm_f(vz[v] + bzz);
          float nn = tanhf(vi[v] + bin + r * (vh[v] + bhn));
          float ho = hF[m * HID + col];
          float hw = (1.0f - z) * nn + z * ho;
          hF[m * HID + col] = hw;
          aB[m * HID + col] = f2bf(hw);
        }
      }
      __syncthreads();
    }

    // ---- RK4 ODE solve: 2 substeps x 4 evals, k-sums kept in registers ----
    {
      float sub = 0.5f * dtS;
#pragma unroll 1
      for (int ss = 0; ss < 2; ++ss) {
        v8f rk0 = {}, rk1 = {};
#pragma unroll 1
        for (int k = 0; k < 4; ++k) {
          layer256(aB, ODE1, p.ode_b1, tB, lane, wave, true); __syncthreads();
          layer256(tB, ODE2, p.ode_b2, cB, lane, wave, true); __syncthreads();
          v8f a0 = {}, a1 = {};
          mm_tile(cB, HID, HID, ODE3, HID, wave * 32, a0, a1, lane);
          const int n0 = wave * 32;
          int n = lane & 15, rb = (lane >> 4) * 8;
          float b30 = p.ode_b3[n0 + n], b31 = p.ode_b3[n0 + 16 + n];
          float w  = (k == 0 || k == 3) ? 1.0f : 2.0f;
          float cc = (k < 2) ? 0.5f * sub : sub;
#pragma unroll
          for (int v = 0; v < 8; ++v) {
            int m = rb + v;
            float k0 = a0[v] + b30, k1 = a1[v] + b31;
            rk0[v] += w * k0;
            rk1[v] += w * k1;
            if (k < 3) {   // stage htmp = h + cc*k for next eval
              aB[m * HID + n0 + n]      = f2bf(hF[m * HID + n0 + n]      + cc * k0);
              aB[m * HID + n0 + 16 + n] = f2bf(hF[m * HID + n0 + 16 + n] + cc * k1);
            }
          }
          __syncthreads();
        }
        const int n0 = wave * 32;
        int n = lane & 15, rb = (lane >> 4) * 8;
        float s6 = sub * (1.0f / 6.0f);
#pragma unroll
        for (int v = 0; v < 8; ++v) {
          int m = rb + v;
          float h0 = hF[m * HID + n0 + n]      + s6 * rk0[v];
          float h1 = hF[m * HID + n0 + 16 + n] + s6 * rk1[v];
          hF[m * HID + n0 + n] = h0;      aB[m * HID + n0 + n]      = f2bf(h0);
          hF[m * HID + n0 + 16 + n] = h1; aB[m * HID + n0 + 16 + n] = f2bf(h1);
        }
        __syncthreads();
      }
    }
  }
}

extern "C" void kernel_launch(void* const* d_in, const int* in_sizes, int n_in,
                              void* d_out, int out_size, void* d_ws, size_t ws_size,
                              hipStream_t stream) {
  (void)in_sizes; (void)n_in; (void)out_size; (void)ws_size;
  const float* inputs = (const float*)d_in[0];
  const float* mask   = (const float*)d_in[1];
  const float* w_ih   = (const float*)d_in[2];
  const float* w_hh   = (const float*)d_in[3];
  const float* b_ih   = (const float*)d_in[4];
  const float* b_hh   = (const float*)d_in[5];
  const float* ode_w1 = (const float*)d_in[6];
  const float* ode_b1 = (const float*)d_in[7];
  const float* ode_w2 = (const float*)d_in[8];
  const float* ode_b2 = (const float*)d_in[9];
  const float* ode_w3 = (const float*)d_in[10];
  const float* ode_b3 = (const float*)d_in[11];
  const float* g_w1   = (const float*)d_in[12];
  const float* g_b1   = (const float*)d_in[13];
  const float* g_w2   = (const float*)d_in[14];
  const float* g_b2   = (const float*)d_in[15];
  const float* g_w3   = (const float*)d_in[16];
  const float* g_b3   = (const float*)d_in[17];
  const float* s_w1   = (const float*)d_in[18];
  const float* s_b1   = (const float*)d_in[19];
  const float* s_w2   = (const float*)d_in[20];
  const float* s_b2   = (const float*)d_in[21];
  const float* s_w3   = (const float*)d_in[22];
  const float* s_b3   = (const float*)d_in[23];

  unsigned short* wts = (unsigned short*)d_ws;
  auto cvt = [&](const float* src, size_t off, int rows, int K, int Kpad) {
    int total = rows * Kpad;
    convert_bf16_pad<<<(total + THREADS - 1) / THREADS, THREADS, 0, stream>>>(
        src, wts + off, rows, K, Kpad);
  };
  cvt(ode_w1, OFF_ODE1, 256, 256, 256);
  cvt(ode_w2, OFF_ODE2, 256, 256, 256);
  cvt(ode_w3, OFF_ODE3, 256, 256, 256);
  cvt(g_w1,   OFF_GW1,  256, 256, 256);
  cvt(g_w2,   OFF_GW2,  256, 256, 256);
  cvt(s_w1,   OFF_SW1,  256, 256, 256);
  cvt(s_w2,   OFF_SW2,  256, 256, 256);
  cvt(s_w3,   OFF_SW3,  64,  256, 256);
  cvt(w_ih,   OFF_WIH,  768, 63,  64);    // pad K 63 -> 64 with zeros
  cvt(w_hh,   OFF_WHH,  768, 256, 256);

  float* out_pred  = (float*)d_out;
  float* out_state = out_pred + (size_t)(TSTEPS - 1) * BTOT;   // 16128 preds, then states

  Params p{ inputs, mask, b_ih, b_hh,
            ode_b1, ode_b2, ode_b3,
            g_b1, g_b2, g_w3, g_b3,
            s_b1, s_b2, s_b3,
            wts, out_pred, out_state };
  liquid_ode_gru<<<BTOT / BTILE, THREADS, 0, stream>>>(p);
}
